// VideoTextAttention_20495583936649
// MI455X (gfx1250) — matrix operand order repaired and checked by reading, not compiled, not run on hardware
//
#include <hip/hip_runtime.h>
#include <hip/hip_bf16.h>

// ---------- CDNA5 WMMA types ----------
typedef __attribute__((ext_vector_type(16))) __bf16 v16bf;
typedef __attribute__((ext_vector_type(8)))  float  v8f;
typedef __attribute__((ext_vector_type(4)))  unsigned int u32x4;

union BFrag {
  u32x4 u[2];   // two 16-byte chunks
  v16bf b;      // 16 bf16 lanes-worth for one wave fragment
};

__device__ __forceinline__ unsigned short f2bf(float x) {
  unsigned int u = __float_as_uint(x);
  unsigned int r = u + 0x7fffu + ((u >> 16) & 1u);   // round-to-nearest-even
  return (unsigned short)(r >> 16);
}

// ---------- problem constants (from reference) ----------
#define BB   16      // batch
#define S1   512     // video seq
#define S2   40      // text seq
#define D1   1024    // video dim (= 4*AD, concat width)
#define D2   300     // text dim
#define AD   256     // attn dim
#define OD   500     // output dim
#define ODP  512     // output dim padded to 16

// ============================================================
// prep: f[b,t,a] = sum_d text[b,t,d] * w2[d,a]   (f32, tiny GEMM)
// ============================================================
__global__ void prep_f_kernel(const float* __restrict__ text,
                              const float* __restrict__ w2,
                              float* __restrict__ f) {
  __shared__ float tr[D2];
  int row = blockIdx.x;           // b*S2 + t  (640 rows)
  int a   = threadIdx.x;          // 0..255
  for (int d = threadIdx.x; d < D2; d += blockDim.x) tr[d] = text[(size_t)row * D2 + d];
  __syncthreads();
  float acc = 0.f;
  for (int d = 0; d < D2; ++d) acc += tr[d] * w2[(size_t)d * AD + a];
  f[(size_t)row * AD + a] = acc;
}

// ============================================================
// prep: w1t[a][k] = bf16(w1[k][a])   (256 x 1024, K-contiguous)
// ============================================================
__global__ void prep_w1t_kernel(const float* __restrict__ w1,
                                unsigned short* __restrict__ w1t) {
  int idx = blockIdx.x * blockDim.x + threadIdx.x;   // 0..262143
  int a = idx >> 10;
  int k = idx & 1023;
  w1t[idx] = f2bf(w1[(size_t)k * AD + a]);
}

// ============================================================
// prep: w4t[n][k] = bf16(w4[k][n]), zero-padded n in [500,512)
// ============================================================
__global__ void prep_w4t_kernel(const float* __restrict__ w4,
                                unsigned short* __restrict__ w4t) {
  int idx = blockIdx.x * blockDim.x + threadIdx.x;   // 0..524287
  int n = idx >> 10;
  int k = idx & 1023;
  float v = (n < OD) ? w4[(size_t)k * OD + n] : 0.f;
  w4t[idx] = f2bf(v);
}

// ============================================================
// fused main kernel: one block = (batch b, 16 rows of s1)
//   e-GEMM (WMMA bf16) -> tanh-attention (TRANS) -> softmax
//   -> text_attn + concat (bf16 LDS) -> out-GEMM (WMMA bf16)
// ============================================================
__global__ __launch_bounds__(256)
void fused_attn_kernel(const float* __restrict__ video,
                       const float* __restrict__ f,
                       const unsigned short* __restrict__ w1t,
                       const unsigned short* __restrict__ w4t,
                       const float* __restrict__ w3,
                       const float* __restrict__ bias,
                       const float* __restrict__ b4,
                       float* __restrict__ out) {
  __shared__ unsigned short ldsAC[16 * D1];  // 32KB: video tile bf16, later cont tile bf16
  __shared__ float ldsE[16 * AD];            // 16KB: e tile f32
  __shared__ float ldsW3[AD];
  __shared__ float ldsBias[AD];
  __shared__ float ldsAt[16 * S2];           // logits -> attn weights

  const int blk  = blockIdx.x;
  const int b    = blk >> 5;              // /32 tiles per batch row dim
  const int s0   = (blk & 31) << 4;       // *16
  const int tid  = threadIdx.x;
  const int lane = tid & 31;
  const int wid  = tid >> 5;              // 8 wave32s

  const int row    = lane & 15;           // A-frag: M row
  const int khalfA = (lane >> 4) << 3;    // A-frag lane-half K offset (0 / 8)
  const int khalfB = (lane >> 4) << 3;    // B-frag lane-half K offset (0 / 8)
  const int rb     = (lane >> 4) << 3;    // C/D: M = v + (lane>=16 ? 8 : 0)

  // ---- stage video tile (16 x 1024 f32 -> bf16 LDS), w3/bias -> LDS ----
  {
    const float* vbase = video + (size_t)(b * S1 + s0) * D1;
    for (int i = tid; i < 16 * D1; i += 256) ldsAC[i] = f2bf(vbase[i]);
    if (tid < AD) { ldsW3[tid] = w3[tid]; ldsBias[tid] = bias[tid]; }
  }
  __syncthreads();

  // ---- e = Avideo(16x1024) x w1(1024x256), WMMA bf16, 2 N-tiles/wave ----
  for (int nt = wid * 2; nt < wid * 2 + 2; ++nt) {
    const int n = nt * 16 + (lane & 15);
    const unsigned short* bcol = w1t + (size_t)n * D1;
    v8f acc = {};
    for (int k0 = 0; k0 < D1; k0 += 32) {
      BFrag af, bg;
      af.u[0] = *(const u32x4*)(&ldsAC[row * D1 + k0 + khalfA]);
      af.u[1] = *(const u32x4*)(&ldsAC[row * D1 + k0 + khalfA + 16]);
      bg.u[0] = *(const u32x4*)(&bcol[k0 + khalfB]);
      bg.u[1] = *(const u32x4*)(&bcol[k0 + khalfB + 16]);
      acc = __builtin_amdgcn_wmma_f32_16x16x32_bf16(false, af.b, false, bg.b,
                                                    (short)0, acc, false, false);
    }
    #pragma unroll
    for (int v = 0; v < 8; ++v) ldsE[(rb + v) * AD + n] = acc[v];
  }
  __syncthreads();

  // ---- logits[s,t] = sum_a w3[a] * tanh(e[s,a] + f[b,t,a] + bias[a]) ----
  for (int idx = tid; idx < 16 * S2; idx += 256) {
    const int s = idx / S2;
    const int t = idx - s * S2;
    const float* frow = f + (size_t)(b * S2 + t) * AD;
    float acc = 0.f;
    for (int a = 0; a < AD; ++a)
      acc += ldsW3[a] * tanhf(ldsE[s * AD + a] + frow[a] + ldsBias[a]);
    ldsAt[s * S2 + t] = acc;
  }
  __syncthreads();

  // ---- softmax over t (one thread per row; 16 rows x 40) ----
  if (tid < 16) {
    const int s = tid;
    float m = -1e30f;
    for (int t = 0; t < S2; ++t) m = fmaxf(m, ldsAt[s * S2 + t]);
    float sum = 0.f;
    for (int t = 0; t < S2; ++t) {
      float ex = __expf(ldsAt[s * S2 + t] - m);
      ldsAt[s * S2 + t] = ex;
      sum += ex;
    }
    const float inv = 1.f / sum;
    for (int t = 0; t < S2; ++t) ldsAt[s * S2 + t] *= inv;
  }
  __syncthreads();

  // ---- text_attn + concat -> tanh -> bf16 cont tile (reuses ldsAC) ----
  for (int idx = tid; idx < 16 * AD; idx += 256) {
    const int s = idx >> 8;
    const int a = idx & 255;
    const float* fb = f + (size_t)b * S2 * AD + a;
    float ta = 0.f;
    #pragma unroll 4
    for (int t = 0; t < S2; ++t) ta += ldsAt[s * S2 + t] * fb[(size_t)t * AD];
    const float e = ldsE[s * AD + a];
    unsigned short* crow = &ldsAC[s * D1];
    crow[a]          = f2bf(tanhf(e));
    crow[AD + a]     = f2bf(tanhf(ta));
    crow[2 * AD + a] = f2bf(tanhf(e * ta));
    crow[3 * AD + a] = f2bf(tanhf(e - ta));
  }
  __syncthreads();

  // ---- out = cont(16x1024) x w4(1024x500), WMMA bf16, 4 N-tiles/wave ----
  for (int nt = wid * 4; nt < wid * 4 + 4; ++nt) {
    const int n = nt * 16 + (lane & 15);
    const unsigned short* bcol = w4t + (size_t)n * D1;
    v8f acc = {};
    for (int k0 = 0; k0 < D1; k0 += 32) {
      BFrag af, bg;
      af.u[0] = *(const u32x4*)(&ldsAC[row * D1 + k0 + khalfA]);
      af.u[1] = *(const u32x4*)(&ldsAC[row * D1 + k0 + khalfA + 16]);
      bg.u[0] = *(const u32x4*)(&bcol[k0 + khalfB]);
      bg.u[1] = *(const u32x4*)(&bcol[k0 + khalfB + 16]);
      acc = __builtin_amdgcn_wmma_f32_16x16x32_bf16(false, af.b, false, bg.b,
                                                    (short)0, acc, false, false);
    }
    if (n < OD) {
      const float bn = b4[n];
      #pragma unroll
      for (int v = 0; v < 8; ++v)
        out[(size_t)(b * S1 + s0 + rb + v) * OD + n] = acc[v] + bn;
    }
  }
}

// ============================================================
// launch
// ============================================================
extern "C" void kernel_launch(void* const* d_in, const int* in_sizes, int n_in,
                              void* d_out, int out_size, void* d_ws, size_t ws_size,
                              hipStream_t stream) {
  const float* video = (const float*)d_in[0];
  const float* text  = (const float*)d_in[1];
  const float* w1    = (const float*)d_in[2];
  const float* w2    = (const float*)d_in[3];
  const float* w3    = (const float*)d_in[4];
  const float* bias  = (const float*)d_in[5];
  const float* w4    = (const float*)d_in[6];
  const float* b4    = (const float*)d_in[7];
  float* out = (float*)d_out;

  // workspace layout (bytes):
  //   f    : BB*S2*AD f32           = 655360
  //   w1t  : AD*D1 bf16             = 524288
  //   w4t  : ODP*D1 bf16            = 1048576
  char* ws = (char*)d_ws;
  float*          f_ws = (float*)ws;
  unsigned short* w1t  = (unsigned short*)(ws + 655360);
  unsigned short* w4t  = (unsigned short*)(ws + 655360 + 524288);

  prep_f_kernel  <<<BB * S2,            256, 0, stream>>>(text, w2, f_ws);
  prep_w1t_kernel<<<(AD * D1) / 256,    256, 0, stream>>>(w1, w1t);
  prep_w4t_kernel<<<(ODP * D1) / 256,   256, 0, stream>>>(w4, w4t);
  fused_attn_kernel<<<BB * (S1 / 16),   256, 0, stream>>>(video, f_ws, w1t, w4t,
                                                          w3, bias, b4, out);
}
